// Cross_Attention_22179211116670
// MI455X (gfx1250) — compile-verified, matrix operand order checked by reading, not run on hardware
//
#include <hip/hip_runtime.h>
#include <hip/hip_bf16.h>
#include <math.h>

// Problem constants (reference: B=8, N=2048, DIM=768)
#define BB 8
#define NN 2048
#define DD 768

typedef __attribute__((ext_vector_type(16))) _Float16 v16h;
typedef __attribute__((ext_vector_type(8)))  _Float16 v8h;
typedef __attribute__((ext_vector_type(8)))  float    v8f;

// ---------------------------------------------------------------------------
// WMMA fragment loaders for V_WMMA_F32_16X16X32_F16 (wave32).
// ISA 16-bit A-matrix 16x32 layout: lane L -> m = L&15, hi = L>>4.
//   frag element j (0..7)  holds K = hi*8 + j        (contiguous!)
//   frag element j (8..15) holds K = 16 + hi*8 + j-8 (contiguous!)
// => a row-major (K-contiguous) fragment is two 16-byte loads + a shuffle.
// B-matrix 32x16 mirrors this with roles (row<->col) swapped.
// All sources below are K-contiguous by construction (transposed producers).
// ---------------------------------------------------------------------------

// frag[row][K] = p[row*ld + K], K contiguous. p must be 16B aligned at tile.
__device__ __forceinline__ v16h frag_rm_h(const _Float16* p, int ld, int row, int hi) {
  const _Float16* r = p + (size_t)row * ld + hi * 8;
  v8h a = *(const v8h*)(r);
  v8h b = *(const v8h*)(r + 16);
  return __builtin_shufflevector(a, b, 0, 1, 2, 3, 4, 5, 6, 7,
                                 8, 9, 10, 11, 12, 13, 14, 15);
}

// same, f32 source converted to f16 with packed converts. 32B aligned.
__device__ __forceinline__ v16h frag_rm_f(const float* p, int ld, int row, int hi) {
  const float* r = p + (size_t)row * ld + hi * 8;
  v8f a = *(const v8f*)(r);
  v8f b = *(const v8f*)(r + 16);
  v8h ah = __builtin_convertvector(a, v8h);
  v8h bh = __builtin_convertvector(b, v8h);
  return __builtin_shufflevector(ah, bh, 0, 1, 2, 3, 4, 5, 6, 7,
                                 8, 9, 10, 11, 12, 13, 14, 15);
}

__device__ __forceinline__ v8f wmma16x16x32(v16h a, v16h b, v8f c) {
  return __builtin_amdgcn_wmma_f32_16x16x32_f16(false, a, false, b, (short)0, c,
                                                false, false);
}

__device__ __forceinline__ float sigmoid1(const float* w) {
  return 1.0f / (1.0f + expf(-w[0]));
}

// ---------------------------------------------------------------------------
// Row softmax (axis=-1) with transposed f16 output At[B, D, N].
// Block = 256 threads handles 32 consecutive rows (n) of one batch.
// Stats per row by one wave (shfl reduction), then 32x32 LDS tile transpose
// so both global reads and global writes are coalesced.
// ---------------------------------------------------------------------------
__global__ void row_softmax_t_f16(const float* __restrict__ x,
                                  _Float16* __restrict__ At,
                                  int B, int N, int D) {
  __shared__ float s_max[32];
  __shared__ float s_inv[32];
  __shared__ _Float16 tile[32][33];  // [n-in-block][d-in-chunk], padded

  int rowBlocks = N / 32;
  int b  = blockIdx.x / rowBlocks;
  int n0 = (blockIdx.x % rowBlocks) * 32;
  const float* xb = x + (size_t)b * N * D;
  _Float16*    Ab = At + (size_t)b * D * N;

  int tid  = threadIdx.x;
  int wave = tid >> 5, lane = tid & 31;

  // stats: 8 waves x 4 rows each
  for (int r = 0; r < 4; ++r) {
    int row = wave * 4 + r;
    const float* rp = xb + (size_t)(n0 + row) * D;
    float m = -1e30f;
    for (int d = lane; d < D; d += 32) m = fmaxf(m, rp[d]);
#pragma unroll
    for (int off = 16; off; off >>= 1) m = fmaxf(m, __shfl_xor(m, off, 32));
    float s = 0.f;
    for (int d = lane; d < D; d += 32) s += expf(rp[d] - m);
#pragma unroll
    for (int off = 16; off; off >>= 1) s += __shfl_xor(s, off, 32);
    if (lane == 0) { s_max[row] = m; s_inv[row] = 1.f / s; }
  }
  __syncthreads();

  int tx = tid & 31;   // inner index (d on read, n on write)
  int ty = tid >> 5;   // 0..7
  for (int d0 = 0; d0 < D; d0 += 32) {
#pragma unroll
    for (int s = 0; s < 4; ++s) {
      int row = ty + 8 * s;
      float v = xb[(size_t)(n0 + row) * D + d0 + tx];
      tile[row][tx] = (_Float16)(expf(v - s_max[row]) * s_inv[row]);
    }
    __syncthreads();
#pragma unroll
    for (int s = 0; s < 4; ++s) {
      int dd = ty + 8 * s;
      Ab[(size_t)(d0 + dd) * N + n0 + tx] = tile[tx][dd];
    }
    __syncthreads();
  }
}

// ---------------------------------------------------------------------------
// Column softmax (axis=1) with transposed f16 output Qt[B, D, N].
// One thread per (b,d) column: online (max,sum) pass, then a contiguous
// write of the whole column (it *is* a row of Qt).
// ---------------------------------------------------------------------------
__global__ void col_softmax_t_f16(const float* __restrict__ x,
                                  _Float16* __restrict__ Qt,
                                  int B, int N, int D) {
  int idx = blockIdx.x * blockDim.x + threadIdx.x;
  if (idx >= B * D) return;
  int b = idx / D, d = idx % D;
  const float* base = x + (size_t)b * N * D + d;
  float m = -1e30f, s = 0.f;
  for (int n = 0; n < N; ++n) {
    float v = base[(size_t)n * D];
    if (v > m) { s = s * expf(m - v) + 1.f; m = v; }
    else       { s += expf(v - m); }
  }
  float inv = 1.f / s;
  _Float16* ob = Qt + (size_t)b * D * N + (size_t)d * N;
  for (int n = 0; n < N; ++n)
    ob[n] = (_Float16)(expf(base[(size_t)n * D] - m) * inv);
}

__global__ void cvt_f32_f16(const float* __restrict__ a,
                            _Float16* __restrict__ o, int n) {
  int i = blockIdx.x * blockDim.x + threadIdx.x;
  if (i < n) o[i] = (_Float16)a[i];
}

// ---------------------------------------------------------------------------
// P[b](d,e) = sum_n At[b](d,n) * Qt[b](e,n)   (D x D, K = N). One tile/wave.
// Both fragments are K-contiguous -> pure b128 loads.
// ---------------------------------------------------------------------------
__global__ void gemm_atb(const _Float16* __restrict__ At,
                         const _Float16* __restrict__ Qt,
                         _Float16* __restrict__ P, int N, int D) {
  int wid   = (blockIdx.x * blockDim.x + threadIdx.x) >> 5;
  int lane  = threadIdx.x & 31;
  int m     = lane & 15, hi = lane >> 4;
  int tiles = D / 16;
  int b  = wid / (tiles * tiles);
  int t  = wid % (tiles * tiles);
  int tm = t / tiles, tn = t % tiles;
  const _Float16* Ab = At + (size_t)b * D * N + (size_t)(tm * 16) * N;
  const _Float16* Qb = Qt + (size_t)b * D * N + (size_t)(tn * 16) * N;
  v8f acc = {};
  for (int k0 = 0; k0 < N; k0 += 32) {
    v16h fa = frag_rm_h(Ab + k0, N, m, hi);
    v16h fb = frag_rm_h(Qb + k0, N, m, hi);
    acc = wmma16x16x32(fa, fb, acc);
  }
  // C layout: VGPR r, lanes 0-15 -> (M=r, N=lane); lanes 16-31 -> (M=r+8).
  _Float16* Pb = P + (size_t)b * D * D + (size_t)(tm * 16 + hi * 8) * D + tn * 16 + m;
#pragma unroll
  for (int r = 0; r < 8; ++r) Pb[(size_t)r * D] = (_Float16)acc[r];
}

// ---------------------------------------------------------------------------
// St[b] = [ f * (P2[b] @ W1^T + P3[b] @ W2^T) ]^T   stored f16, transposed:
// each lane's 8 accumulator rows are contiguous in St -> single b128 store.
// ---------------------------------------------------------------------------
__global__ void gemm_s(const _Float16* __restrict__ P2,
                       const _Float16* __restrict__ P3,
                       const _Float16* __restrict__ W1h,
                       const _Float16* __restrict__ W2h,
                       const float* __restrict__ w,
                       _Float16* __restrict__ St, int D) {
  int wid   = (blockIdx.x * blockDim.x + threadIdx.x) >> 5;
  int lane  = threadIdx.x & 31;
  int m     = lane & 15, hi = lane >> 4;
  int tiles = D / 16;
  int b  = wid / (tiles * tiles);
  int t  = wid % (tiles * tiles);
  int tm = t / tiles, tn = t % tiles;
  float f = sigmoid1(w);
  const _Float16* P2b = P2 + (size_t)b * D * D + (size_t)(tm * 16) * D;
  const _Float16* P3b = P3 + (size_t)b * D * D + (size_t)(tm * 16) * D;
  const _Float16* W1b = W1h + (size_t)(tn * 16) * D;  // W^T[k][e] = W[e][k]
  const _Float16* W2b = W2h + (size_t)(tn * 16) * D;
  v8f acc = {};
  for (int k0 = 0; k0 < D; k0 += 32) {
    acc = wmma16x16x32(frag_rm_h(P2b + k0, D, m, hi),
                       frag_rm_h(W1b + k0, D, m, hi), acc);
    acc = wmma16x16x32(frag_rm_h(P3b + k0, D, m, hi),
                       frag_rm_h(W2b + k0, D, m, hi), acc);
  }
  // St[e][d] = S[d][e]; lane holds e = tn*16+m, d = tm*16+hi*8 + r (contig).
  v8h sv;
#pragma unroll
  for (int r = 0; r < 8; ++r) sv[r] = (_Float16)(f * acc[r]);
  *(v8h*)(St + (size_t)b * D * D + (size_t)(tn * 16 + m) * D + tm * 16 + hi * 8) = sv;
}

// ---------------------------------------------------------------------------
// out[b] = x[b] @ S[b] + f*(b1+b2) + x[b]   (N x D, K = D)
// A from x (f32 -> packed cvt to f16); B from St row-major along K.
// ---------------------------------------------------------------------------
__global__ void gemm_out(const float* __restrict__ x,
                         const _Float16* __restrict__ St,
                         const float* __restrict__ b1,
                         const float* __restrict__ b2,
                         const float* __restrict__ w,
                         float* __restrict__ out, int N, int D) {
  int wid    = (blockIdx.x * blockDim.x + threadIdx.x) >> 5;
  int lane   = threadIdx.x & 31;
  int m      = lane & 15, hi = lane >> 4;
  int tilesN = N / 16, tilesD = D / 16;
  int b  = wid / (tilesN * tilesD);
  int t  = wid % (tilesN * tilesD);
  int tm = t / tilesD, tn = t % tilesD;
  float f = sigmoid1(w);
  const float*    xb = x + (size_t)b * N * D + (size_t)(tm * 16) * D;
  const _Float16* Sb = St + (size_t)b * D * D + (size_t)(tn * 16) * D;
  v8f acc = {};
  for (int k0 = 0; k0 < D; k0 += 32) {
    if (k0 + 32 < D) __builtin_prefetch(xb + k0 + 32, 0, 0);
    // B[k][e] = S[k][e] = St[e][k] -> K-contiguous in St.
    acc = wmma16x16x32(frag_rm_f(xb + k0, D, m, hi),
                       frag_rm_h(Sb + k0, D, m, hi), acc);
  }
  int   col  = tn * 16 + m;
  float bias = f * (b1[col] + b2[col]);
  const float* xrow = x + (size_t)b * N * D;
  float*       ob   = out + (size_t)b * N * D;
  int   row0 = tm * 16 + hi * 8;
#pragma unroll
  for (int r = 0; r < 8; ++r) {
    size_t idx = (size_t)(row0 + r) * D + col;
    ob[idx] = acc[r] + bias + xrow[idx];
  }
}

// ---------------------------------------------------------------------------
// Host launcher.  Inputs: x, x2, x3, W1, b1, W2, b2, w (setup_inputs order).
// Workspace layout (~128 MB), all f16 blocks 32B-aligned:
//   A2t,Q2t,A3t,Q3t : f16 [B,D,N] transposed softmaxes   4 * 25.17 MB
//   P2,P3           : f16 [B,D,D]                        2 *  9.44 MB
//   W1h,W2h         : f16 [D,D]                          2 *  1.18 MB
//   St              : f16 [B,D,D] (S transposed)              9.44 MB
// ---------------------------------------------------------------------------
extern "C" void kernel_launch(void* const* d_in, const int* in_sizes, int n_in,
                              void* d_out, int out_size, void* d_ws, size_t ws_size,
                              hipStream_t stream) {
  (void)in_sizes; (void)n_in; (void)out_size; (void)ws_size;
  const float* x  = (const float*)d_in[0];
  const float* x2 = (const float*)d_in[1];
  const float* x3 = (const float*)d_in[2];
  const float* W1 = (const float*)d_in[3];
  const float* b1 = (const float*)d_in[4];
  const float* W2 = (const float*)d_in[5];
  const float* b2 = (const float*)d_in[6];
  const float* w  = (const float*)d_in[7];
  float* out = (float*)d_out;

  const size_t BND = (size_t)BB * NN * DD;
  const size_t BDD = (size_t)BB * DD * DD;
  const size_t DDs = (size_t)DD * DD;

  char* p = (char*)d_ws;
  _Float16* A2t = (_Float16*)p; p += BND * sizeof(_Float16);
  _Float16* Q2t = (_Float16*)p; p += BND * sizeof(_Float16);
  _Float16* A3t = (_Float16*)p; p += BND * sizeof(_Float16);
  _Float16* Q3t = (_Float16*)p; p += BND * sizeof(_Float16);
  _Float16* P2  = (_Float16*)p; p += BDD * sizeof(_Float16);
  _Float16* P3  = (_Float16*)p; p += BDD * sizeof(_Float16);
  _Float16* W1h = (_Float16*)p; p += DDs * sizeof(_Float16);
  _Float16* W2h = (_Float16*)p; p += DDs * sizeof(_Float16);
  _Float16* St  = (_Float16*)p;

  // 1) softmaxes -> transposed f16 tensors
  {
    dim3 g(BB * NN / 32), blk(256);  // 512 blocks
    row_softmax_t_f16<<<g, blk, 0, stream>>>(x2, A2t, BB, NN, DD);
    row_softmax_t_f16<<<g, blk, 0, stream>>>(x3, A3t, BB, NN, DD);
  }
  {
    int cols = BB * DD;  // 6144 threads
    dim3 g((cols + 255) / 256), blk(256);
    col_softmax_t_f16<<<g, blk, 0, stream>>>(x2, Q2t, BB, NN, DD);
    col_softmax_t_f16<<<g, blk, 0, stream>>>(x3, Q3t, BB, NN, DD);
  }
  // 2) weights to f16
  {
    int n = DD * DD;
    dim3 g((n + 255) / 256), blk(256);
    cvt_f32_f16<<<g, blk, 0, stream>>>(W1, W1h, n);
    cvt_f32_f16<<<g, blk, 0, stream>>>(W2, W2h, n);
  }
  // 3) P_i = A^T Q  (8 * 48 * 48 = 18432 waves)
  {
    int waves = BB * (DD / 16) * (DD / 16);
    dim3 g(waves * 32 / 256), blk(256);
    gemm_atb<<<g, blk, 0, stream>>>(A2t, Q2t, P2, NN, DD);
    gemm_atb<<<g, blk, 0, stream>>>(A3t, Q3t, P3, NN, DD);
  }
  // 4) St = (f*(P2 W1^T + P3 W2^T))^T  as f16
  {
    int waves = BB * (DD / 16) * (DD / 16);
    dim3 g(waves * 32 / 256), blk(256);
    gemm_s<<<g, blk, 0, stream>>>(P2, P3, W1h, W2h, w, St, DD);
  }
  // 5) out = x @ S + f*(b1+b2) + x   (8 * 128 * 48 = 49152 waves)
  {
    int waves = BB * (NN / 16) * (DD / 16);
    dim3 g(waves * 32 / 256), blk(256);
    gemm_out<<<g, blk, 0, stream>>>(x, St, b1, b2, w, out, NN, DD);
  }
}
